// AttentionCritic_14010183319671
// MI455X (gfx1250) — compile-verified, matrix operand order checked by reading, not run on hardware
//
#include <hip/hip_runtime.h>
#include <math.h>

typedef unsigned short u16;
typedef unsigned int u32;
typedef unsigned long long u64;

typedef __attribute__((ext_vector_type(16))) __bf16 v16bf;
typedef __attribute__((ext_vector_type(8)))  float  v8f;

// ---------------- problem constants ----------------
#define NAG 4           // agents
#define BATCH 32768     // batch
#define SDIM 512        // state dim
#define ADIM 64         // action dim
#define HDIM 256        // hidden
#define NHEAD 16
#define DHEAD 16
#define SA_DIM 576      // SDIM + ADIM

// ---------------- bf16 helpers ----------------
__device__ __forceinline__ u16 f2bf(float f) {
    u32 u = __float_as_uint(f);
    u32 r = u + 0x7FFFu + ((u >> 16) & 1u);   // round-to-nearest-even
    return (u16)(r >> 16);
}
__device__ __forceinline__ float bf2f(u16 h) {
    return __uint_as_float(((u32)h) << 16);
}

union FragB16 { uint4 q[2]; v16bf v; };

// async 16B global -> LDS copy (GVS mode: SGPR base + per-lane i32 offset)
__device__ __forceinline__ void async_ld16(u32 lds_off, u32 goff, u64 base) {
    asm volatile("global_load_async_to_lds_b128 %0, %1, %2"
                 :: "v"(lds_off), "v"(goff), "s"(base) : "memory");
}

// =====================================================================
// 1) BatchNorm statistics: partial sums (coalesced column tiles)
// grid: (NAG, 18, 16) block: 256   (8 row-groups x 32 cols)
// =====================================================================
__global__ __launch_bounds__(256) void stats_partial_kernel(
    const float* __restrict__ states, const float* __restrict__ actions,
    float* __restrict__ psum, float* __restrict__ psq)
{
    const int n = blockIdx.x;
    const int f = blockIdx.y * 32 + (threadIdx.x & 31);
    const int g = threadIdx.x >> 5;           // row group 0..7
    const int z = blockIdx.z;                 // row chunk 0..15

    const float* src; int ld;
    if (f < SDIM) { src = states  + (long)n * BATCH * SDIM + f; ld = SDIM; }
    else          { src = actions + (long)n * BATCH * ADIM + (f - SDIM); ld = ADIM; }

    float s = 0.f, q = 0.f;
    const int r0 = z * 2048;
    for (int r = r0 + g; r < r0 + 2048; r += 8) {
        float x = src[(long)r * ld];
        s += x; q += x * x;
    }
    __shared__ float S[256], Q[256];
    S[threadIdx.x] = s; Q[threadIdx.x] = q;
    __syncthreads();
    if (g == 0) {
        const int c = threadIdx.x & 31;
        #pragma unroll
        for (int gg = 1; gg < 8; ++gg) { s += S[gg * 32 + c]; q += Q[gg * 32 + c]; }
        long o = ((long)z * NAG + n) * SA_DIM + f;
        psum[o] = s; psq[o] = q;
    }
}

__global__ __launch_bounds__(256) void stats_final_kernel(
    const float* __restrict__ psum, const float* __restrict__ psq,
    float* __restrict__ mean, float* __restrict__ rstd)
{
    int t = blockIdx.x * 256 + threadIdx.x;
    if (t >= NAG * SA_DIM) return;
    int n = t / SA_DIM, f = t % SA_DIM;
    float s = 0.f, q = 0.f;
    #pragma unroll
    for (int z = 0; z < 16; ++z) {
        long o = ((long)z * NAG + n) * SA_DIM + f;
        s += psum[o]; q += psq[o];
    }
    float m = s * (1.f / BATCH);
    float v = q * (1.f / BATCH) - m * m;
    mean[t] = m;
    rstd[t] = rsqrtf(v + 1e-5f);
}

// =====================================================================
// 2) weight prep: fp32 [K][Nt] per agent -> bf16 TRANSPOSED [Nt][K]
//    (so LDS B-tiles are contiguous 64B runs -> async-copyable)
// =====================================================================
__global__ __launch_bounds__(256) void cvt_transpose_kernel(
    const float* __restrict__ src, u16* __restrict__ dst,
    int K, int Nt, long total)
{
    long i = (long)blockIdx.x * blockDim.x + threadIdx.x;
    long stride = (long)gridDim.x * blockDim.x;
    const long mat = (long)K * Nt;
    for (; i < total; i += stride) {
        long a = i / mat;
        long rem = i - a * mat;
        int n = (int)(rem / K);
        int k = (int)(rem - (long)n * K);
        dst[i] = f2bf(src[a * mat + (long)k * Nt + n]);
    }
}

// src: [NHEAD][HDIM][DHEAD] -> dst transposed: [NHEAD*DHEAD][HDIM]
__global__ __launch_bounds__(256) void remap_head_kernel(
    const float* __restrict__ src, u16* __restrict__ dst)
{
    int t = blockIdx.x * 256 + threadIdx.x;      // 65536 threads
    int c = t >> 8;          // output row  = e*16+d
    int h = t & 255;         // output col  = hidden index
    int e = c >> 4, d = c & 15;
    dst[t] = f2bf(src[((e << 8) + h) * DHEAD + d]);
}

// =====================================================================
// 3) Tiled bf16 WMMA GEMM.  Block tile 128(M) x 128(N) x 32(K), 8 waves,
//    wave tile 32x64 = 2x4 v_wmma_f32_16x16x32_bf16 tiles.
//    Weights arrive pre-transposed [Nout][K]; staged via async->LDS.
// AMODE: 0 = fp32 concat(states,actions) + batchnorm (sync A stage)
//        1 = fp32 single source + batchnorm       (sync A stage)
//        2 = bf16 activations: dual-buffered fully-async A+W staging
// =====================================================================
template<int AMODE, bool LRELU, bool OUTF32>
__global__ __launch_bounds__(256) void gemm_kernel(
    const float* __restrict__ srcS, long srcS_ag, int ldS,
    const float* __restrict__ srcT, long srcT_ag, int ldT,
    const u16*  __restrict__ abf,  long abf_ag,  int lda,
    const float* __restrict__ mean, const float* __restrict__ rstd, int stats_ag,
    const u16*  __restrict__ w,    long w_ag,
    const float* __restrict__ bias, int bias_ag,
    u16* __restrict__ outb, float* __restrict__ outf, long out_ag, int ldd,
    int K, int Nout)
{
    const int z    = blockIdx.z;
    const int tid  = threadIdx.x;
    const int lane = tid & 31;
    const int wid  = tid >> 5;
    const int waveM = wid & 3;       // 0..3 -> 32-row slice
    const int waveN = wid >> 2;      // 0..1 -> 64-col slice

    __shared__ u16 lA[2][128 * 40];     // [row][k] stride 40 (padded)
    __shared__ u16 lBT[2][128 * 40];    // [col][k] stride 40

    v8f zero = {};
    v8f acc[2][4];
    #pragma unroll
    for (int i = 0; i < 2; ++i)
        #pragma unroll
        for (int j = 0; j < 4; ++j) acc[i][j] = zero;

    const long rowbase = (long)blockIdx.x * 128;
    const int  colbase = blockIdx.y * 128;
    const int  nkt = K >> 5;

    const u64 wpv = (u64)(size_t)(w + (long)z * w_ag);
    const u64 apv = (AMODE == 2) ? (u64)(size_t)(abf + (long)z * abf_ag) : 0;

    // ---- async stagers (4 async ops / thread / tile in AMODE2; 2 in BN) ----
    auto issueW = [&](int kt, int bb) {
        #pragma unroll
        for (int p = 0; p < 2; ++p) {
            int cc  = tid + 256 * p;       // 0..511 : 128 rows x 4 chunks
            int col = cc >> 2;
            int kp  = (cc & 3) * 8;
            int colg = colbase + col;
            if (colg >= Nout) colg = Nout - 1;     // clamp (masked at store)
            u32 ldso = (u32)(size_t)&lBT[bb][col * 40 + kp];
            u32 goff = (u32)((((long)colg * K) + kt * 32 + kp) * 2);
            async_ld16(ldso, goff, wpv);
        }
    };
    auto issueA = [&](int kt, int bb) {
        #pragma unroll
        for (int p = 0; p < 2; ++p) {
            int cc  = tid + 256 * p;
            int row = cc >> 2;
            int kp  = (cc & 3) * 8;
            u32 ldso = (u32)(size_t)&lA[bb][row * 40 + kp];
            u32 goff = (u32)(((rowbase + row) * (long)lda + kt * 32 + kp) * 2);
            async_ld16(ldso, goff, apv);
        }
    };
    // ---- WMMA on one LDS buffer ----
    auto do_tile = [&](int bb) {
        const int kh = (lane >> 4) << 3;     // 0 or 8 (ISA bf16 A/B layout)
        FragB16 af[2], bfr[4];
        #pragma unroll
        for (int tr = 0; tr < 2; ++tr) {
            int m = waveM * 32 + tr * 16 + (lane & 15);
            af[tr].q[0] = *(const uint4*)&lA[bb][m * 40 + kh];
            af[tr].q[1] = *(const uint4*)&lA[bb][m * 40 + 16 + kh];
        }
        #pragma unroll
        for (int tc = 0; tc < 4; ++tc) {
            int nn = waveN * 64 + tc * 16 + (lane & 15);
            bfr[tc].q[0] = *(const uint4*)&lBT[bb][nn * 40 + kh];
            bfr[tc].q[1] = *(const uint4*)&lBT[bb][nn * 40 + 16 + kh];
        }
        #pragma unroll
        for (int tr = 0; tr < 2; ++tr)
            #pragma unroll
            for (int tc = 0; tc < 4; ++tc)
                acc[tr][tc] = __builtin_amdgcn_wmma_f32_16x16x32_bf16(
                    false, af[tr].v, false, bfr[tc].v,
                    (short)0, acc[tr][tc], false, false);
    };

    if constexpr (AMODE == 2) {
        // -------- dual-buffered fully-async pipeline --------
        issueA(0, 0); issueW(0, 0);
        for (int kt = 0; kt < nkt; ++kt) {
            const int cur = kt & 1;
            if (kt + 1 < nkt) {
                issueA(kt + 1, cur ^ 1); issueW(kt + 1, cur ^ 1);
                asm volatile("s_wait_asynccnt 0x4" ::: "memory");
            } else {
                asm volatile("s_wait_asynccnt 0x0" ::: "memory");
            }
            __syncthreads();
            do_tile(cur);
            __syncthreads();
        }
    } else {
        // -------- batchnorm-fused A stage (sync) + async W stage --------
        for (int kt = 0; kt < nkt; ++kt) {
            const int kg = kt * 32;
            issueW(kt, 0);
            {
                const float* src; int ld; int colb;
                if (AMODE == 0 && kg >= SDIM) {
                    src = srcT + (long)z * srcT_ag; ld = ldT; colb = kg - SDIM;
                } else {
                    src = srcS + (long)z * srcS_ag; ld = ldS; colb = kg;
                }
                const float* mn = mean + (long)z * stats_ag;
                const float* rs = rstd + (long)z * stats_ag;
                const int c4 = (tid & 7) * 4;
                float m0 = mn[kg + c4 + 0], m1 = mn[kg + c4 + 1];
                float m2 = mn[kg + c4 + 2], m3 = mn[kg + c4 + 3];
                float r0 = rs[kg + c4 + 0], r1 = rs[kg + c4 + 1];
                float r2 = rs[kg + c4 + 2], r3 = rs[kg + c4 + 3];
                #pragma unroll
                for (int p = 0; p < 4; ++p) {
                    int r = p * 32 + (tid >> 3);
                    float4 x = *(const float4*)(src + (rowbase + r) * ld + colb + c4);
                    u32 lo = (u32)f2bf((x.x - m0) * r0) | ((u32)f2bf((x.y - m1) * r1) << 16);
                    u32 hi = (u32)f2bf((x.z - m2) * r2) | ((u32)f2bf((x.w - m3) * r3) << 16);
                    *(uint2*)&lA[0][r * 40 + c4] = make_uint2(lo, hi);
                }
            }
            asm volatile("s_wait_asynccnt 0x0" ::: "memory");
            __syncthreads();
            do_tile(0);
            __syncthreads();
        }
    }

    // ---- epilogue: bias + leaky relu + store ----
    const float* bb = bias ? (bias + (long)z * bias_ag) : nullptr;
    #pragma unroll
    for (int tr = 0; tr < 2; ++tr) {
        const long mb = rowbase + waveM * 32 + tr * 16 + ((lane >> 4) << 3);
        #pragma unroll
        for (int tc = 0; tc < 4; ++tc) {
            const int nb = colbase + waveN * 64 + tc * 16 + (lane & 15);
            if (nb < Nout) {
                float bv = bb ? bb[nb] : 0.f;
                #pragma unroll
                for (int r = 0; r < 8; ++r) {
                    float v = acc[tr][tc][r] + bv;
                    if (LRELU) v = (v > 0.f) ? v : 0.01f * v;
                    long idx = (long)z * out_ag + (mb + r) * ldd + nb;
                    if (OUTF32) outf[idx] = v;
                    else        outb[idx] = f2bf(v);
                }
            }
        }
    }
}

// =====================================================================
// 4) attention across agents: thread per (batch, head)
// =====================================================================
__device__ __forceinline__ void unp16(uint4 a, uint4 b, float* f) {
    u32 u[8] = { a.x, a.y, a.z, a.w, b.x, b.y, b.z, b.w };
    #pragma unroll
    for (int i = 0; i < 8; ++i) {
        f[2 * i]     = bf2f((u16)(u[i] & 0xffffu));
        f[2 * i + 1] = bf2f((u16)(u[i] >> 16));
    }
}

__global__ __launch_bounds__(256) void attn_kernel(
    const u16* __restrict__ keys, const u16* __restrict__ vals,
    const u16* __restrict__ sel,  u16* __restrict__ cin)
{
    const long t = (long)blockIdx.x * 256 + threadIdx.x;   // < BATCH*NHEAD
    const int  b = (int)(t >> 4);
    const int  e = (int)(t & 15);
    const long off = ((long)b << 8) + (e << 4);
    const long AG  = (long)BATCH * 256;
    const long AGC = (long)BATCH * 512;

    float kf[NAG][16];
    #pragma unroll
    for (int j = 0; j < NAG; ++j) {
        const u16* p = keys + (long)j * AG + off;
        unp16(*(const uint4*)p, *(const uint4*)(p + 8), kf[j]);
    }

    float prob[NAG][NAG];
    #pragma unroll
    for (int i = 0; i < NAG; ++i) {
        float sf[16];
        const u16* p = sel + (long)i * AG + off;
        unp16(*(const uint4*)p, *(const uint4*)(p + 8), sf);
        float lg[NAG];
        #pragma unroll
        for (int j = 0; j < NAG; ++j) {
            float d = 0.f;
            #pragma unroll
            for (int q = 0; q < 16; ++q) d += sf[q] * kf[j][q];
            lg[j] = d * 0.25f;                     // / sqrt(16)
        }
        lg[i] = -1e9f;                             // self mask
        float mx = lg[0];
        #pragma unroll
        for (int j = 1; j < NAG; ++j) mx = fmaxf(mx, lg[j]);
        float sum = 0.f;
        #pragma unroll
        for (int j = 0; j < NAG; ++j) { prob[i][j] = __expf(lg[j] - mx); sum += prob[i][j]; }
        float inv = 1.f / sum;
        #pragma unroll
        for (int j = 0; j < NAG; ++j) prob[i][j] *= inv;
    }

    #pragma unroll
    for (int j = 0; j < NAG; ++j) {                 // reuse kf for values
        const u16* p = vals + (long)j * AG + off;
        unp16(*(const uint4*)p, *(const uint4*)(p + 8), kf[j]);
    }

    #pragma unroll
    for (int i = 0; i < NAG; ++i) {
        float o[16];
        #pragma unroll
        for (int q = 0; q < 16; ++q) {
            float s = 0.f;
            #pragma unroll
            for (int j = 0; j < NAG; ++j) s += prob[i][j] * kf[j][q];
            o[q] = s;
        }
        u32 u[8];
        #pragma unroll
        for (int q = 0; q < 8; ++q)
            u[q] = (u32)f2bf(o[2 * q]) | ((u32)f2bf(o[2 * q + 1]) << 16);
        u16* dst = cin + (long)i * AGC + ((long)b << 9) + 256 + (e << 4);
        *(uint4*)dst       = make_uint4(u[0], u[1], u[2], u[3]);
        *(uint4*)(dst + 8) = make_uint4(u[4], u[5], u[6], u[7]);
    }
}

// =====================================================================
// 5) argmax(actions) gather from all_q
// =====================================================================
__global__ __launch_bounds__(256) void gather_kernel(
    const float* __restrict__ actions, const float* __restrict__ allq,
    float* __restrict__ out)
{
    const long t = (long)blockIdx.x * 256 + threadIdx.x;   // < NAG*BATCH
    const float4* ap = (const float4*)(actions + t * ADIM);
    float best = -INFINITY; int bidx = 0;
    #pragma unroll
    for (int i = 0; i < 16; ++i) {
        float4 x = ap[i];
        if (x.x > best) { best = x.x; bidx = 4 * i; }
        if (x.y > best) { best = x.y; bidx = 4 * i + 1; }
        if (x.z > best) { best = x.z; bidx = 4 * i + 2; }
        if (x.w > best) { best = x.w; bidx = 4 * i + 3; }
    }
    out[t] = allq[t * ADIM + bidx];
}

// =====================================================================
// host launcher
// =====================================================================
extern "C" void kernel_launch(void* const* d_in, const int* in_sizes, int n_in,
                              void* d_out, int out_size, void* d_ws, size_t ws_size,
                              hipStream_t stream) {
    const float* states  = (const float*)d_in[0];
    const float* actions = (const float*)d_in[1];
    const float* Wenc    = (const float*)d_in[2];
    const float* benc    = (const float*)d_in[3];
    const float* Ws      = (const float*)d_in[4];
    const float* bs      = (const float*)d_in[5];
    const float* Wk      = (const float*)d_in[6];
    const float* Wsel    = (const float*)d_in[7];
    const float* Wv      = (const float*)d_in[8];
    const float* bv      = (const float*)d_in[9];
    const float* Wc1     = (const float*)d_in[10];
    const float* bc1     = (const float*)d_in[11];
    const float* Wc2     = (const float*)d_in[12];
    const float* bc2     = (const float*)d_in[13];
    float* out = (float*)d_out;

    char* ws = (char*)d_ws;
    constexpr size_t OFF_MEAN   = 0;          // 4*576 f32
    constexpr size_t OFF_RSTD   = 16384;
    constexpr size_t OFF_PSUM   = 32768;      // 16*4*576 f32
    constexpr size_t OFF_PSQ    = 196608;
    constexpr size_t OFF_WENC   = 393216;     // bf16 4*[256][576] (transposed)
    constexpr size_t OFF_WS     = 1572864;    // bf16 4*[256][512]
    constexpr size_t OFF_WK     = 2621440;    // bf16 [256][256]
    constexpr size_t OFF_WSEL   = 2752512;
    constexpr size_t OFF_WV     = 2883584;
    constexpr size_t OFF_WC1    = 3014656;    // bf16 4*[256][512]
    constexpr size_t OFF_WC2    = 4063232;    // bf16 4*[64][256]
    constexpr size_t OFF_SAENC  = 4194304;    // bf16 4*B*256   (64MB)
    constexpr size_t OFF_KEYS   = OFF_SAENC + 67108864;
    constexpr size_t OFF_VALS   = OFF_KEYS  + 67108864;
    constexpr size_t OFF_SEL    = OFF_VALS  + 67108864;
    constexpr size_t OFF_CIN    = OFF_SEL   + 67108864;   // bf16 4*B*512 (128MB)
    constexpr size_t OFF_H      = OFF_CIN   + 134217728;  // bf16 4*B*256
    constexpr size_t OFF_ALLQ   = OFF_H     + 67108864;   // f32  4*B*64  (32MB)

    float* mean = (float*)(ws + OFF_MEAN);
    float* rstd = (float*)(ws + OFF_RSTD);
    float* psum = (float*)(ws + OFF_PSUM);
    float* psq  = (float*)(ws + OFF_PSQ);
    u16* wenc_b = (u16*)(ws + OFF_WENC);
    u16* ws_b   = (u16*)(ws + OFF_WS);
    u16* wk_b   = (u16*)(ws + OFF_WK);
    u16* wsel_b = (u16*)(ws + OFF_WSEL);
    u16* wv_b   = (u16*)(ws + OFF_WV);
    u16* wc1_b  = (u16*)(ws + OFF_WC1);
    u16* wc2_b  = (u16*)(ws + OFF_WC2);
    u16* sa_enc = (u16*)(ws + OFF_SAENC);
    u16* keys   = (u16*)(ws + OFF_KEYS);
    u16* vals   = (u16*)(ws + OFF_VALS);
    u16* sel    = (u16*)(ws + OFF_SEL);
    u16* cin    = (u16*)(ws + OFF_CIN);
    u16* hbuf   = (u16*)(ws + OFF_H);
    float* allq = (float*)(ws + OFF_ALLQ);

    const long AG_H  = (long)BATCH * 256;
    const long AG_C  = (long)BATCH * 512;

    // 1) batchnorm stats
    stats_partial_kernel<<<dim3(NAG, 18, 16), 256, 0, stream>>>(states, actions, psum, psq);
    stats_final_kernel<<<9, 256, 0, stream>>>(psum, psq, mean, rstd);

    // 2) weight prep (fp32 -> bf16, transposed [Nout][K])
    cvt_transpose_kernel<<<512, 256, 0, stream>>>(Wenc, wenc_b, SA_DIM, HDIM, (long)NAG * SA_DIM * HDIM);
    cvt_transpose_kernel<<<512, 256, 0, stream>>>(Ws,   ws_b,   SDIM,   HDIM, (long)NAG * SDIM * HDIM);
    cvt_transpose_kernel<<<512, 256, 0, stream>>>(Wc1,  wc1_b,  512,    HDIM, (long)NAG * 512 * HDIM);
    cvt_transpose_kernel<<<256, 256, 0, stream>>>(Wc2,  wc2_b,  HDIM,   ADIM, (long)NAG * HDIM * ADIM);
    remap_head_kernel<<<256, 256, 0, stream>>>(Wk,   wk_b);
    remap_head_kernel<<<256, 256, 0, stream>>>(Wsel, wsel_b);
    remap_head_kernel<<<256, 256, 0, stream>>>(Wv,   wv_b);

    // 3) sa_enc = lrelu(bn(concat(s,a)) @ Wenc + benc)        [bf16, ld 256]
    gemm_kernel<0, true, false><<<dim3(256, 2, NAG), 256, 0, stream>>>(
        states, (long)BATCH * SDIM, SDIM,  actions, (long)BATCH * ADIM, ADIM,
        nullptr, 0, 0,  mean, rstd, SA_DIM,
        wenc_b, (long)SA_DIM * HDIM,  benc, HDIM,
        sa_enc, nullptr, AG_H, 256,  SA_DIM, HDIM);

    // 4) s_enc = lrelu(bn(states) @ Ws + bs)  -> cin[:, 0:256] [bf16, ld 512]
    gemm_kernel<1, true, false><<<dim3(256, 2, NAG), 256, 0, stream>>>(
        states, (long)BATCH * SDIM, SDIM,  nullptr, 0, 0,
        nullptr, 0, 0,  mean, rstd, SA_DIM,
        ws_b, (long)SDIM * HDIM,  bs, HDIM,
        cin, nullptr, AG_C, 512,  SDIM, HDIM);

    // 5) keys / vals / sel  (shared head weights, agent stride 0)
    gemm_kernel<2, false, false><<<dim3(256, 2, NAG), 256, 0, stream>>>(
        nullptr, 0, 0, nullptr, 0, 0,
        sa_enc, AG_H, 256,  nullptr, nullptr, 0,
        wk_b, 0,  nullptr, 0,
        keys, nullptr, AG_H, 256,  HDIM, HDIM);
    gemm_kernel<2, true, false><<<dim3(256, 2, NAG), 256, 0, stream>>>(
        nullptr, 0, 0, nullptr, 0, 0,
        sa_enc, AG_H, 256,  nullptr, nullptr, 0,
        wv_b, 0,  bv, 0,
        vals, nullptr, AG_H, 256,  HDIM, HDIM);
    gemm_kernel<2, false, false><<<dim3(256, 2, NAG), 256, 0, stream>>>(
        nullptr, 0, 0, nullptr, 0, 0,
        cin, AG_C, 512,  nullptr, nullptr, 0,
        wsel_b, 0,  nullptr, 0,
        sel, nullptr, AG_H, 256,  HDIM, HDIM);

    // 6) attention across agents -> cin[:, 256:512]
    attn_kernel<<<(BATCH * NHEAD) / 256, 256, 0, stream>>>(keys, vals, sel, cin);

    // 7) h = lrelu(cin @ Wc1 + bc1)                            [bf16, ld 256]
    gemm_kernel<2, true, false><<<dim3(256, 2, NAG), 256, 0, stream>>>(
        nullptr, 0, 0, nullptr, 0, 0,
        cin, AG_C, 512,  nullptr, nullptr, 0,
        wc1_b, (long)512 * HDIM,  bc1, HDIM,
        hbuf, nullptr, AG_H, 256,  512, HDIM);

    // 8) all_q = h @ Wc2 + bc2                                 [f32, ld 64]
    gemm_kernel<2, false, true><<<dim3(256, 1, NAG), 256, 0, stream>>>(
        nullptr, 0, 0, nullptr, 0, 0,
        hbuf, AG_H, 256,  nullptr, nullptr, 0,
        wc2_b, (long)HDIM * ADIM,  bc2, ADIM,
        nullptr, allq, (long)BATCH * ADIM, ADIM,  HDIM, ADIM);

    // 9) q = all_q[argmax(actions)]
    gather_kernel<<<(NAG * BATCH) / 256, 256, 0, stream>>>(actions, allq, out);
}
